// ADIAModel_60756607369327
// MI455X (gfx1250) — compile-verified, matrix operand order checked by reading, not run on hardware
//
#include <hip/hip_runtime.h>

// ---------------------------------------------------------------------------
// Types / constants
// ---------------------------------------------------------------------------
typedef _Float16 half_t;
typedef __attribute__((ext_vector_type(16))) _Float16 v16h;
typedef __attribute__((ext_vector_type(8)))  _Float16 v8h;
typedef __attribute__((ext_vector_type(8)))  float    v8f;

constexpr int B_    = 32;
constexpr int E_    = 132;
constexpr int BE    = B_ * E_;       // 4224
constexpr int D_    = 64;
constexpr int HEADS = 4;
constexpr int HD    = 16;
constexpr int NCLS  = 8;

static __device__ __forceinline__ float gelu_f(float x) {
    return 0.5f * x * (1.0f + erff(x * 0.70710678118654752440f));
}

static __device__ __forceinline__ v16h cat8(v8h lo, v8h hi) {
    return __builtin_shufflevector(lo, hi, 0,1,2,3,4,5,6,7,8,9,10,11,12,13,14,15);
}

static __device__ __forceinline__ v8f wmma_f16(v16h a, v16h b, v8f c) {
    return __builtin_amdgcn_wmma_f32_16x16x32_f16(false, a, false, b,
                                                  (short)0, c, false, false);
}

// fragment k-index pattern: lane group g, half h
static __device__ __forceinline__ int frag_k(int kc, int g, int h) {
    return kc * 32 + ((h < 8) ? (g * 8 + h) : (16 + g * 8 + (h - 8)));
}

// ---------------------------------------------------------------------------
// Weight packing into per-lane WMMA B-fragment order (branchless gather).
// ---------------------------------------------------------------------------
__global__ void pack_w(const float* __restrict__ src, half_t* __restrict__ dst,
                       int K, int N, int Kch, int Ntiles, int conv) {
    int total = Ntiles * Kch * 512;
    for (int t = blockIdx.x * blockDim.x + threadIdx.x; t < total;
         t += gridDim.x * blockDim.x) {
        int h    = t & 15;
        int lane = (t >> 4) & 31;
        int kc   = (t >> 9) % Kch;
        int nt   = t / (Kch * 512);
        int n    = nt * 16 + (lane & 15);
        int g    = lane >> 4;
        int k    = frag_k(kc, g, h);
        bool ok  = (k < K) && (n < N);
        size_t idx = ok ? (conv ? (size_t)n * K + k : (size_t)k * N + n) : 0;
        float v = src[idx];
        dst[t] = (half_t)(ok ? v : 0.0f);
    }
}

// ---------------------------------------------------------------------------
// FUSED conv1 (8->32, 32x32, s1, GELU) + conv2 (32->32, ->16x16, s2).
// One 8-wave block per image. conv1's full 64 KB output lives only in LDS;
// conv2 consumes it via branchless ds gathers. Saves ~550 MB of HBM traffic.
// All gathers are unconditional loads from clamped addresses + v_cndmask so
// the 16 loads per fragment pipeline under a single counter wait.
// ---------------------------------------------------------------------------
__global__ __launch_bounds__(256)
void conv12_fused(const float* __restrict__ inp, const half_t* __restrict__ w1pk,
                  const float* __restrict__ b1, const half_t* __restrict__ w2pk,
                  const float* __restrict__ b2, half_t* __restrict__ act2out) {
    __shared__ __align__(16) half_t A1[32 * 1024];   // conv1 out [oc][pix], 64 KB

    const int be   = blockIdx.x;
    const int wave = threadIdx.x >> 5, lane = threadIdx.x & 31;
    const int g = lane >> 4, ml = lane & 15;
    const float* img = inp + (size_t)be * 8 * 1024;  // [8][32][32] f32, L1/L2-hot

    // ---------------- conv1: 64 pixel-tiles (16 px x 32 oc each) ------------
    for (int t = wave; t < 64; t += 8) {
        const int pix = t * 16 + ml;          // this lane's A row (pixel)
        const int oy = pix >> 5, ox = pix & 31;
        v8f c0 = {}, c1 = {};
        for (int kc = 0; kc < 3; ++kc) {      // K = 72 padded to 96
            v16h a;
            #pragma unroll
            for (int h = 0; h < 16; ++h) {
                int k  = frag_k(kc, g, h);
                int ic = k / 9, r9 = k % 9, kh = r9 / 3, kw = r9 % 3;
                int iy = oy + kh - 1, ix = ox + kw - 1;
                bool ok = (k < 72) & ((unsigned)iy < 32u) & ((unsigned)ix < 32u);
                int idx = ok ? (ic * 1024 + iy * 32 + ix) : 0;
                float v = img[idx];                       // always in-bounds
                a[h] = (half_t)(ok ? v : 0.0f);
            }
            v16h bA = *(const v16h*)(w1pk + (size_t)(0 * 3 + kc) * 512 + lane * 16);
            v16h bB = *(const v16h*)(w1pk + (size_t)(1 * 3 + kc) * 512 + lane * 16);
            c0 = wmma_f16(a, bA, c0);
            c1 = wmma_f16(a, bB, c1);
        }
        const int oc0 = ml, oc1 = 16 + ml;
        const float bv0 = b1[oc0], bv1 = b1[oc1];
        for (int r = 0; r < 8; ++r) {
            int pp = t * 16 + r + g * 8;      // C layout: M = r + 8*(lane>=16)
            A1[oc0 * 1024 + pp] = (half_t)gelu_f(c0[r] + bv0);
            A1[oc1 * 1024 + pp] = (half_t)gelu_f(c1[r] + bv1);
        }
    }
    __syncthreads();

    // ---------------- conv2 from LDS: 16 pixel-tiles of 16x16 output -------
    for (int t = wave; t < 16; t += 8) {
        const int pix = t * 16 + ml;
        const int oy = pix >> 4, ox = pix & 15;
        v8f c0 = {}, c1 = {};
        for (int kc = 0; kc < 9; ++kc) {      // K = 288, no padding
            v16h a;
            #pragma unroll
            for (int h = 0; h < 16; ++h) {
                int k  = frag_k(kc, g, h);
                int ic = k / 9, r9 = k % 9, kh = r9 / 3, kw = r9 % 3;
                int iy = oy * 2 + kh - 1, ix = ox * 2 + kw - 1;
                bool ok = ((unsigned)iy < 32u) & ((unsigned)ix < 32u);
                int idx = ok ? (ic * 1024 + iy * 32 + ix) : 0;
                half_t v = A1[idx];                       // always in-bounds
                a[h] = ok ? v : (half_t)0.0f;
            }
            v16h bA = *(const v16h*)(w2pk + (size_t)(0 * 9 + kc) * 512 + lane * 16);
            v16h bB = *(const v16h*)(w2pk + (size_t)(1 * 9 + kc) * 512 + lane * 16);
            c0 = wmma_f16(a, bA, c0);
            c1 = wmma_f16(a, bB, c1);
        }
        const int oc0 = ml, oc1 = 16 + ml;
        const float bv0 = b2[oc0], bv1 = b2[oc1];
        for (int r = 0; r < 8; ++r) {
            int pp = t * 16 + r + g * 8;
            act2out[((size_t)be * 32 + oc0) * 256 + pp] = (half_t)(c0[r] + bv0);
            act2out[((size_t)be * 32 + oc1) * 256 + pp] = (half_t)(c1[r] + bv1);
        }
    }
}

// ---------------------------------------------------------------------------
// Implicit-GEMM 3x3 conv (pad=1), WMMA f16 -> f32 accum (used for conv3/4;
// their inputs are L2-resident so im2col re-reads are cache-served).
// ---------------------------------------------------------------------------
template<int IC, int OC, int IH, int IW, int S>
__global__ __launch_bounds__(128)
void conv3x3_wmma(const half_t* __restrict__ inp, const half_t* __restrict__ wpk,
                  const float* __restrict__ bias, half_t* __restrict__ out) {
    constexpr int OH   = (S == 1) ? IH : IH / 2;
    constexpr int OW   = (S == 1) ? IW : IW / 2;
    constexpr int K    = IC * 9;
    constexpr int KP   = (K + 31) / 32 * 32;
    constexpr int KCH  = KP / 32;
    constexpr int NT   = OC / 16;
    constexpr int MTB  = 4 / NT;
    constexpr int MBLK = MTB * 16;

    __shared__ __align__(16) half_t As[MBLK][KP];

    const int be      = blockIdx.y;
    const int pixbase = blockIdx.x * MBLK;
    const int tid     = threadIdx.x;
    const half_t* imgb = inp + (size_t)be * IC * IH * IW;

    for (int idx = tid; idx < MBLK * KP; idx += 128) {
        int row = idx / KP, k = idx % KP;
        int pix = pixbase + row;
        int oy = pix / OW, ox = pix % OW;
        int ic = k / 9, r9 = k % 9, kh = r9 / 3, kw = r9 % 3;
        int iy = oy * S + kh - 1, ix = ox * S + kw - 1;
        bool ok = (k < K) & (pix < OH * OW) &
                  ((unsigned)iy < (unsigned)IH) & ((unsigned)ix < (unsigned)IW);
        int src = ok ? (ic * IH * IW + iy * IW + ix) : 0;
        half_t v = imgb[src];                             // always in-bounds
        As[row][k] = ok ? v : (half_t)0.0f;
    }
    __syncthreads();

    const int wave = tid >> 5, lane = tid & 31;
    const int mt = wave / NT, nt = wave % NT;
    const int g = lane >> 4, ml = lane & 15;

    v8f c = {};
    const half_t* wbase = wpk + (size_t)nt * KCH * 512 + lane * 16;
    for (int kc = 0; kc < KCH; ++kc) {
        const half_t* ap = &As[mt * 16 + ml][kc * 32 + g * 8];
        v16h a = cat8(*(const v8h*)ap, *(const v8h*)(ap + 16));
        v16h b = *(const v16h*)(wbase + (size_t)kc * 512);
        c = wmma_f16(a, b, c);
    }

    const int oc = nt * 16 + ml;
    const float bv = bias[oc];
    const int mrowbase = pixbase + mt * 16 + g * 8;
    for (int r = 0; r < 8; ++r) {
        int pix = mrowbase + r;
        if (pix < OH * OW)
            out[((size_t)be * OC + oc) * (OH * OW) + pix] = (half_t)(c[r] + bv);
    }
}

// ---------------------------------------------------------------------------
// Generic WMMA GEMM: C[M,Nreal] = act(A[M,KP] @ Wpk + bias)
// ---------------------------------------------------------------------------
template<int ACTV>   // 0 = none, 1 = GELU
__global__ __launch_bounds__(128)
void gemm_wmma_k(const half_t* __restrict__ A, const half_t* __restrict__ wpk,
                 const float* __restrict__ bias,
                 float* __restrict__ cf, half_t* __restrict__ ch,
                 int Mtiles, int Nreal, int Kch, int lda) {
    const int wave = threadIdx.x >> 5, lane = threadIdx.x & 31;
    const int mt = blockIdx.x * 4 + wave;
    if (mt >= Mtiles) return;
    const int nt = blockIdx.y;
    const int g = lane >> 4, ml = lane & 15;

    const half_t* arow  = A + (size_t)(mt * 16 + ml) * lda;
    const half_t* wbase = wpk + (size_t)nt * Kch * 512 + lane * 16;

    v8f c = {};
    for (int kc = 0; kc < Kch; ++kc) {
        const half_t* ap = arow + kc * 32 + g * 8;
        __builtin_prefetch(ap + 32, 0, 0);              // global_prefetch_b8
        v16h a = cat8(*(const v8h*)ap, *(const v8h*)(ap + 16));
        v16h b = *(const v16h*)(wbase + (size_t)kc * 512);
        c = wmma_f16(a, b, c);
    }

    const int col = nt * 16 + ml;
    if (col < Nreal) {
        float bv = bias ? bias[col] : 0.0f;
        const int mbase = mt * 16 + g * 8;
        for (int r = 0; r < 8; ++r) {
            float v = c[r] + bv;
            if (ACTV) v = gelu_f(v);
            size_t o = (size_t)(mbase + r) * Nreal + col;
            if (cf) cf[o] = v;
            if (ch) ch[o] = (half_t)v;
        }
    }
}

// ---------------------------------------------------------------------------
// GroupNorm (8 groups) stats + in-place normalize(+GELU) on f16 activations
// ---------------------------------------------------------------------------
__global__ void gn_stats(const half_t* __restrict__ x, float* __restrict__ st,
                         int C, int HWp) {
    int bg = blockIdx.x;               // be*8 + group
    int be = bg >> 3, grp = bg & 7;
    int cg = C / 8;
    int n  = cg * HWp;
    const half_t* base = x + ((size_t)be * C + grp * cg) * HWp;
    float s = 0.f, s2 = 0.f;
    for (int i = threadIdx.x; i < n; i += blockDim.x) {
        float v = (float)base[i];
        s += v; s2 += v * v;
    }
    __shared__ float ss[256], sq[256];
    ss[threadIdx.x] = s; sq[threadIdx.x] = s2;
    __syncthreads();
    for (int k = blockDim.x >> 1; k > 0; k >>= 1) {
        if (threadIdx.x < k) { ss[threadIdx.x] += ss[threadIdx.x + k];
                               sq[threadIdx.x] += sq[threadIdx.x + k]; }
        __syncthreads();
    }
    if (threadIdx.x == 0) {
        float m   = ss[0] / n;
        float var = sq[0] / n - m * m;
        st[bg * 2]     = m;
        st[bg * 2 + 1] = rsqrtf(var + 1e-5f);
    }
}

__global__ void gn_apply(half_t* __restrict__ x, const float* __restrict__ st,
                         const float* __restrict__ gmm, const float* __restrict__ bta,
                         int C, int HWp) {
    size_t total = (size_t)BE * C * HWp;
    for (size_t i = (size_t)blockIdx.x * blockDim.x + threadIdx.x; i < total;
         i += (size_t)gridDim.x * blockDim.x) {
        int c  = (int)((i / HWp) % C);
        int be = (int)(i / ((size_t)C * HWp));
        int grp = c / (C / 8);
        float m = st[(be * 8 + grp) * 2];
        float r = st[(be * 8 + grp) * 2 + 1];
        float v = ((float)x[i] - m) * r * gmm[c] + bta[c];
        x[i] = (half_t)gelu_f(v);
    }
}

// mean-pool 4x4 per channel -> f16 [BE][64]
__global__ void pool_k(const half_t* __restrict__ a4, half_t* __restrict__ pooled) {
    int be = blockIdx.x, c = threadIdx.x;   // 64 threads
    const half_t* p = a4 + ((size_t)be * 64 + c) * 16;
    float s = 0.f;
    for (int i = 0; i < 16; ++i) s += (float)p[i];
    pooled[(size_t)be * 64 + c] = (half_t)(s * (1.0f / 16.0f));
}

// build em = concat(conv_emb, type_emb[edge_types]) as f16 [BE][128]
__global__ void embuild(const float* __restrict__ cemb, const int* __restrict__ et,
                        const float* __restrict__ temb, half_t* __restrict__ em) {
    int i = blockIdx.x * blockDim.x + threadIdx.x;  // BE*64
    if (i >= BE * 64) return;
    int row = i >> 6, d = i & 63;
    int t = et[row];
    em[(size_t)row * 128 + d]      = (half_t)cemb[i];
    em[(size_t)row * 128 + 64 + d] = (half_t)temb[(size_t)t * 64 + d];
}

// LayerNorm over width 64, optional residual add, optional GELU; writes f32+f16
template<int GELU_>
__global__ void ln_k(const float* __restrict__ a, const float* __restrict__ badd,
                     const float* __restrict__ gmm, const float* __restrict__ bta,
                     float* __restrict__ of, half_t* __restrict__ oh) {
    int row = blockIdx.x, d = threadIdx.x;   // 64 threads
    float v = a[(size_t)row * 64 + d];
    if (badd) v += badd[(size_t)row * 64 + d];
    __shared__ float red[64];
    __shared__ float mean_s, rstd_s;
    red[d] = v; __syncthreads();
    for (int s = 32; s > 0; s >>= 1) { if (d < s) red[d] += red[d + s]; __syncthreads(); }
    if (d == 0) mean_s = red[0] * (1.0f / 64.0f);
    __syncthreads();
    float dv = v - mean_s;
    red[d] = dv * dv; __syncthreads();
    for (int s = 32; s > 0; s >>= 1) { if (d < s) red[d] += red[d + s]; __syncthreads(); }
    if (d == 0) rstd_s = rsqrtf(red[0] * (1.0f / 64.0f) + 1e-5f);
    __syncthreads();
    float o = dv * rstd_s * gmm[d] + bta[d];
    if (GELU_) o = gelu_f(o);
    if (of) of[(size_t)row * 64 + d] = o;
    if (oh) oh[(size_t)row * 64 + d] = (half_t)o;
}

// ---------------------------------------------------------------------------
// Attention: one block per (batch, head); online softmax over 132 keys with
// structural-relation bias and key mask. Writes f16 concat-head output.
// ---------------------------------------------------------------------------
__global__ __launch_bounds__(128)
void attn_k(const float* __restrict__ q, const float* __restrict__ k,
            const float* __restrict__ v, const float* __restrict__ sb,
            const unsigned char* __restrict__ mask, half_t* __restrict__ outh) {
    int b = blockIdx.x >> 2;
    int h = blockIdx.x & 3;
    __shared__ float Qs[E_][HD], Ks[E_][HD], Vs[E_][HD];
    for (int i = threadIdx.x; i < E_ * HD; i += blockDim.x) {
        int r = i / HD, d = i % HD;
        size_t off = ((size_t)b * E_ + r) * D_ + h * HD + d;
        Qs[r][d] = q[off]; Ks[r][d] = k[off]; Vs[r][d] = v[off];
    }
    __syncthreads();
    const float scale = 0.25f;   // hd^-0.5, hd=16
    for (int r = threadIdx.x; r < E_; r += blockDim.x) {
        int u1 = r / 11, t1 = r % 11;
        int v1 = t1 + (t1 >= u1 ? 1 : 0);
        float qreg[HD];
        for (int d = 0; d < HD; ++d) qreg[d] = Qs[r][d];
        float mx = -3.4e38f, sum = 0.f;
        float acc[HD];
        for (int d = 0; d < HD; ++d) acc[d] = 0.f;
        for (int c = 0; c < E_; ++c) {
            if (!mask[(size_t)b * E_ + c]) continue;   // -inf -> exp 0
            float dot = 0.f;
            for (int d = 0; d < HD; ++d) dot += qreg[d] * Ks[c][d];
            int u2 = c / 11, t2 = c % 11;
            int v2 = t2 + (t2 >= u2 ? 1 : 0);
            int rel;
            if (u1 == v2 && v1 == u2) rel = 0;
            else if (u1 == u2)        rel = 1;
            else if (v1 == v2)        rel = 2;
            else if (v1 == u2)        rel = 3;
            else if (u1 == v2)        rel = 4;
            else                      rel = 5;
            float s = dot * scale + sb[rel * HEADS + h];
            if (s > mx) {
                float f = __expf(mx - s);
                sum = sum * f + 1.0f;
                for (int d = 0; d < HD; ++d) acc[d] = acc[d] * f + Vs[c][d];
                mx = s;
            } else {
                float e = __expf(s - mx);
                sum += e;
                for (int d = 0; d < HD; ++d) acc[d] += e * Vs[c][d];
            }
        }
        float inv = (sum > 0.f) ? 1.0f / sum : 0.0f;   // nan_to_num
        for (int d = 0; d < HD; ++d)
            outh[((size_t)b * E_ + r) * D_ + h * HD + d] = (half_t)(acc[d] * inv);
    }
}

// gather node embeddings: embs[b,j] = concat of 4 selected edge rows
__global__ void gather_k(const half_t* __restrict__ x, half_t* __restrict__ embs) {
    int i = blockIdx.x * blockDim.x + threadIdx.x;   // 320*64
    if (i >= 320 * 64) return;
    int row = i >> 6, d = i & 63;
    int b = row / 10, j = row % 10;
    int o = j + 2;
    int e_vx = o * 11, e_vy = o * 11 + 1, e_xv = o - 1, e_yv = 11 + o - 1;
    size_t base = (size_t)b * E_;
    embs[(size_t)row * 256 + d]       = x[(base + e_vx) * 64 + d];
    embs[(size_t)row * 256 + 64 + d]  = x[(base + e_vy) * 64 + d];
    embs[(size_t)row * 256 + 128 + d] = x[(base + e_xv) * 64 + d];
    embs[(size_t)row * 256 + 192 + d] = x[(base + e_yv) * 64 + d];
}

// ---------------------------------------------------------------------------
// Host orchestration
// ---------------------------------------------------------------------------
extern "C" void kernel_launch(void* const* d_in, const int* in_sizes, int n_in,
                              void* d_out, int out_size, void* d_ws, size_t ws_size,
                              hipStream_t stream) {
    (void)in_sizes; (void)n_in; (void)out_size; (void)ws_size;

    // --- input pointers (setup_inputs dict order, params depth-first) ---
    const float*         edge_data = (const float*)d_in[0];
    const int*           edge_type = (const int*)d_in[1];
    const unsigned char* edge_mask = (const unsigned char*)d_in[2];
    // d_in[3] = p (unused, compile-time constant)
    const float* c1w = (const float*)d_in[4];  const float* c1b = (const float*)d_in[5];
    const float* c2w = (const float*)d_in[6];  const float* c2b = (const float*)d_in[7];
    const float* gn2g = (const float*)d_in[8]; const float* gn2b = (const float*)d_in[9];
    const float* c3w = (const float*)d_in[10]; const float* c3b = (const float*)d_in[11];
    const float* gn3g = (const float*)d_in[12]; const float* gn3b = (const float*)d_in[13];
    const float* c4w = (const float*)d_in[14]; const float* c4b = (const float*)d_in[15];
    const float* gn4g = (const float*)d_in[16]; const float* gn4b = (const float*)d_in[17];
    const float* extw = (const float*)d_in[18]; const float* extb = (const float*)d_in[19];
    const float* temb = (const float*)d_in[20];
    const float* emw  = (const float*)d_in[21]; const float* emb  = (const float*)d_in[22];
    const float* emg  = (const float*)d_in[23]; const float* emb2 = (const float*)d_in[24];
    const float* nmw  = (const float*)d_in[25]; const float* nmb  = (const float*)d_in[26];
    const float* nmg  = (const float*)d_in[27]; const float* nmb2 = (const float*)d_in[28];
    const float* nhw  = (const float*)d_in[29]; const float* nhb  = (const float*)d_in[30];
    const int L0 = 31, L1 = 48;  // layer param bases (17 each)

    // --- workspace arena ---
    char* wsb = (char*)d_ws;
    size_t off = 0;
    auto alloc = [&](size_t n) -> void* {
        off = (off + 255) & ~(size_t)255;
        void* p = wsb + off;
        off += n;
        return p;
    };
    auto hsz = [](size_t n) { return n * sizeof(half_t); };

    // packed weights
    half_t* wc1  = (half_t*)alloc(hsz(2 * 3 * 512));
    half_t* wc2  = (half_t*)alloc(hsz(2 * 9 * 512));
    half_t* wc3  = (half_t*)alloc(hsz(4 * 9 * 512));
    half_t* wc4  = (half_t*)alloc(hsz(4 * 18 * 512));
    half_t* wext = (half_t*)alloc(hsz(4 * 2 * 512));
    half_t* wem  = (half_t*)alloc(hsz(4 * 4 * 512));
    half_t* wnm  = (half_t*)alloc(hsz(4 * 8 * 512));
    half_t* wnh  = (half_t*)alloc(hsz(1 * 2 * 512));
    half_t* wqp[2]; half_t* wkp[2]; half_t* wvp[2]; half_t* wop[2];
    half_t* w1p[2]; half_t* w2p[2];
    for (int l = 0; l < 2; ++l) {
        wqp[l] = (half_t*)alloc(hsz(4 * 2 * 512));
        wkp[l] = (half_t*)alloc(hsz(4 * 2 * 512));
        wvp[l] = (half_t*)alloc(hsz(4 * 2 * 512));
        wop[l] = (half_t*)alloc(hsz(4 * 2 * 512));
        w1p[l] = (half_t*)alloc(hsz(16 * 2 * 512));
        w2p[l] = (half_t*)alloc(hsz(4 * 8 * 512));
    }

    // activations / buffers (no act1 in HBM: it lives in LDS of the fused kernel)
    half_t* act2   = (half_t*)alloc(hsz((size_t)BE * 32 * 256));
    half_t* act3   = (half_t*)alloc(hsz((size_t)BE * 64 * 64));
    half_t* act4   = (half_t*)alloc(hsz((size_t)BE * 64 * 16));
    float*  stats  = (float*)alloc((size_t)BE * 8 * 2 * sizeof(float));
    half_t* pooled = (half_t*)alloc(hsz((size_t)BE * 64));
    float*  cemb   = (float*)alloc((size_t)BE * 64 * sizeof(float));
    half_t* em     = (half_t*)alloc(hsz((size_t)BE * 128));
    float*  emmm   = (float*)alloc((size_t)BE * 64 * sizeof(float));
    float*  xf     = (float*)alloc((size_t)BE * 64 * sizeof(float));
    half_t* xh     = (half_t*)alloc(hsz((size_t)BE * 64));
    float*  qb     = (float*)alloc((size_t)BE * 64 * sizeof(float));
    float*  kb     = (float*)alloc((size_t)BE * 64 * sizeof(float));
    float*  vb     = (float*)alloc((size_t)BE * 64 * sizeof(float));
    half_t* aoh    = (half_t*)alloc(hsz((size_t)BE * 64));
    float*  ob     = (float*)alloc((size_t)BE * 64 * sizeof(float));
    half_t* f1h    = (half_t*)alloc(hsz((size_t)BE * 256));
    float*  f2f    = (float*)alloc((size_t)BE * 64 * sizeof(float));
    half_t* embs   = (half_t*)alloc(hsz((size_t)320 * 256));
    float*  mm     = (float*)alloc((size_t)320 * 64 * sizeof(float));
    half_t* mh     = (half_t*)alloc(hsz((size_t)320 * 64));

    // --- pack all weights ---
    auto PK = [&](const float* src, half_t* dst, int K, int N, int conv) {
        int Kch = (K + 31) / 32, Nt = (N + 15) / 16;
        int total = Nt * Kch * 512;
        pack_w<<<(total + 255) / 256, 256, 0, stream>>>(src, dst, K, N, Kch, Nt, conv);
    };
    PK(c1w, wc1, 72, 32, 1);   PK(c2w, wc2, 288, 32, 1);
    PK(c3w, wc3, 288, 64, 1);  PK(c4w, wc4, 576, 64, 1);
    PK(extw, wext, 64, 64, 0); PK(emw, wem, 128, 64, 0);
    PK(nmw, wnm, 256, 64, 0);  PK(nhw, wnh, 64, 8, 0);
    for (int l = 0; l < 2; ++l) {
        int base = l ? L1 : L0;
        PK((const float*)d_in[base + 0],  wqp[l], 64, 64, 0);
        PK((const float*)d_in[base + 2],  wkp[l], 64, 64, 0);
        PK((const float*)d_in[base + 4],  wvp[l], 64, 64, 0);
        PK((const float*)d_in[base + 6],  wop[l], 64, 64, 0);
        PK((const float*)d_in[base + 11], w1p[l], 64, 256, 0);
        PK((const float*)d_in[base + 13], w2p[l], 256, 64, 0);
    }

    // --- conv stack: fused conv1+conv2 (LDS-resident act1), then conv3/4 ---
    conv12_fused<<<BE, 256, 0, stream>>>(edge_data, wc1, c1b, wc2, c2b, act2);
    gn_stats<<<BE * 8, 256, 0, stream>>>(act2, stats, 32, 256);
    {   size_t tot = (size_t)BE * 32 * 256;
        gn_apply<<<(unsigned)((tot + 255) / 256), 256, 0, stream>>>(act2, stats, gn2g, gn2b, 32, 256); }
    conv3x3_wmma<32, 64, 16, 16, 2>
        <<<dim3(4, BE), 128, 0, stream>>>(act2, wc3, c3b, act3);
    gn_stats<<<BE * 8, 256, 0, stream>>>(act3, stats, 64, 64);
    {   size_t tot = (size_t)BE * 64 * 64;
        gn_apply<<<(unsigned)((tot + 255) / 256), 256, 0, stream>>>(act3, stats, gn3g, gn3b, 64, 64); }
    conv3x3_wmma<64, 64, 8, 8, 2>
        <<<dim3(1, BE), 128, 0, stream>>>(act3, wc4, c4b, act4);
    gn_stats<<<BE * 8, 256, 0, stream>>>(act4, stats, 64, 16);
    {   size_t tot = (size_t)BE * 64 * 16;
        gn_apply<<<(unsigned)((tot + 255) / 256), 256, 0, stream>>>(act4, stats, gn4g, gn4b, 64, 16); }

    // --- pool + ext projection + edge embedding ---
    pool_k<<<BE, 64, 0, stream>>>(act4, pooled);
    const int MT = BE / 16;  // 264
    gemm_wmma_k<0><<<dim3((MT + 3) / 4, 4), 128, 0, stream>>>(
        pooled, wext, extb, cemb, nullptr, MT, 64, 2, 64);
    embuild<<<(BE * 64 + 255) / 256, 256, 0, stream>>>(cemb, edge_type, temb, em);
    gemm_wmma_k<0><<<dim3((MT + 3) / 4, 4), 128, 0, stream>>>(
        em, wem, emb, emmm, nullptr, MT, 64, 4, 128);
    ln_k<1><<<BE, 64, 0, stream>>>(emmm, nullptr, emg, emb2, xf, xh);

    // --- 2 transformer layers ---
    for (int l = 0; l < 2; ++l) {
        int base = l ? L1 : L0;
        const float* bq  = (const float*)d_in[base + 1];
        const float* bk  = (const float*)d_in[base + 3];
        const float* bv  = (const float*)d_in[base + 5];
        const float* bo  = (const float*)d_in[base + 7];
        const float* sb  = (const float*)d_in[base + 8];
        const float* g1  = (const float*)d_in[base + 9];
        const float* b1  = (const float*)d_in[base + 10];
        const float* bf1 = (const float*)d_in[base + 12];
        const float* bf2 = (const float*)d_in[base + 14];
        const float* g2  = (const float*)d_in[base + 15];
        const float* b2  = (const float*)d_in[base + 16];

        gemm_wmma_k<0><<<dim3((MT + 3) / 4, 4), 128, 0, stream>>>(
            xh, wqp[l], bq, qb, nullptr, MT, 64, 2, 64);
        gemm_wmma_k<0><<<dim3((MT + 3) / 4, 4), 128, 0, stream>>>(
            xh, wkp[l], bk, kb, nullptr, MT, 64, 2, 64);
        gemm_wmma_k<0><<<dim3((MT + 3) / 4, 4), 128, 0, stream>>>(
            xh, wvp[l], bv, vb, nullptr, MT, 64, 2, 64);
        attn_k<<<B_ * HEADS, 128, 0, stream>>>(qb, kb, vb, sb, edge_mask, aoh);
        gemm_wmma_k<0><<<dim3((MT + 3) / 4, 4), 128, 0, stream>>>(
            aoh, wop[l], bo, ob, nullptr, MT, 64, 2, 64);
        ln_k<0><<<BE, 64, 0, stream>>>(xf, ob, g1, b1, xf, xh);
        gemm_wmma_k<1><<<dim3((MT + 3) / 4, 16), 128, 0, stream>>>(
            xh, w1p[l], bf1, nullptr, f1h, MT, 256, 2, 64);
        gemm_wmma_k<0><<<dim3((MT + 3) / 4, 4), 128, 0, stream>>>(
            f1h, w2p[l], bf2, f2f, nullptr, MT, 64, 8, 256);
        ln_k<0><<<BE, 64, 0, stream>>>(xf, f2f, g2, b2, xf, xh);
    }

    // --- node merge + head ---
    gather_k<<<(320 * 64 + 255) / 256, 256, 0, stream>>>(xh, embs);
    gemm_wmma_k<0><<<dim3(5, 4), 128, 0, stream>>>(
        embs, wnm, nmb, mm, nullptr, 20, 64, 8, 256);
    ln_k<1><<<320, 64, 0, stream>>>(mm, nullptr, nmg, nmb2, nullptr, mh);
    gemm_wmma_k<0><<<dim3(5, 1), 128, 0, stream>>>(
        mh, wnh, nhb, (float*)d_out, nullptr, 20, NCLS, 2, 64);
}
